// Attention_20392504721713
// MI455X (gfx1250) — matrix operand order repaired and checked by reading, not compiled, not run on hardware
//
#include <hip/hip_runtime.h>
#include <hip/hip_bf16.h>

typedef __bf16 bf16;
typedef __attribute__((ext_vector_type(16))) __bf16 v16bf;
typedef __attribute__((ext_vector_type(8)))  float  v8f;

#define WMMA_BF16(a, b, c) \
  __builtin_amdgcn_wmma_f32_16x16x32_bf16(false, (a), false, (b), (short)0, (c), false, false)

// ---- CDNA5 async global->LDS copy (ASYNCcnt-tracked), guarded for safety ----
#if __has_builtin(__builtin_amdgcn_global_load_async_to_lds_b128)
#define HAVE_ASYNC_LDS 1
typedef int v4i_ __attribute__((ext_vector_type(4)));
typedef __attribute__((address_space(1))) v4i_* as1_v4i_p;  // global <4 x i32>*
typedef __attribute__((address_space(3))) v4i_* as3_v4i_p;  // LDS    <4 x i32>*
#define ASYNC_COPY_B128(g, l)                                                  \
  __builtin_amdgcn_global_load_async_to_lds_b128((as1_v4i_p)(g), (as3_v4i_p)(l), 0, 0)
#if __has_builtin(__builtin_amdgcn_s_wait_asynccnt)
#define ASYNC_WAIT() __builtin_amdgcn_s_wait_asynccnt(0)
#else
#define ASYNC_WAIT() asm volatile("s_wait_asynccnt 0x0" ::: "memory")
#endif
#endif

// ---------------------------------------------------------------------------
// f32 -> bf16 elementwise conversion, 4 elements/thread (v_cvt_pk_bf16_f32)
// ---------------------------------------------------------------------------
__global__ __launch_bounds__(256) void cvt_f32_to_bf16_x4(const float* __restrict__ src,
                                                          bf16* __restrict__ dst, int n4) {
  int i = blockIdx.x * blockDim.x + threadIdx.x;
  if (i < n4) {
    float4 f = ((const float4*)src)[i];
    bf16 o[4];
    o[0] = (bf16)f.x; o[1] = (bf16)f.y; o[2] = (bf16)f.z; o[3] = (bf16)f.w;
    *(uint2*)&dst[i * 4] = *(uint2*)o;
  }
}

// ---------------------------------------------------------------------------
// WMMA GEMM: C[M,N] = A[M,K] * Bw[K,N] + bias[N]
// A, Bw bf16 row-major; output either f32 (Cf) or bf16 (Cb).
// Block: 128 threads (4 waves, 2x2 grid), tile 128x128, K-step 32.
// Each wave: 64x64 tile = 4x4 accumulators = 16 wmma per K-step.
// A tile staged via async global->LDS; weight tile staged TRANSPOSED so both
// fragment classes are contiguous ds_load_b128s.
// ---------------------------------------------------------------------------
__global__ __launch_bounds__(128) void gemm_wmma_bf16(const bf16* __restrict__ A,
                                                      const bf16* __restrict__ Bw,
                                                      const float* __restrict__ bias,
                                                      float* __restrict__ Cf,
                                                      bf16* __restrict__ Cb,
                                                      int M, int N, int K) {
  constexpr int BM = 128, BN = 128, BK = 32;
  constexpr int LDA  = BK + 8;  // 40 elems = 80 B rows (16B aligned), padded
  constexpr int LDBT = BK + 8;
  __shared__ __align__(16) bf16 As[BM * LDA];     // 10240 B
  __shared__ __align__(16) bf16 Bst[BN * LDBT];   // 10240 B, transposed [n][k]

  const int tid  = threadIdx.x;
  const int lane = tid & 31, wave = tid >> 5;
  const int half = lane >> 4, lm = lane & 15;
  const int wm = wave >> 1, wn = wave & 1;  // 2x2 wave grid
  const int mo = wm * 64, no = wn * 64;
  const int bm0 = blockIdx.y * BM, bn0 = blockIdx.x * BN;

  v8f acc[4][4];
#pragma unroll
  for (int mi = 0; mi < 4; mi++)
#pragma unroll
    for (int ni = 0; ni < 4; ni++)
#pragma unroll
      for (int e = 0; e < 8; e++) acc[mi][ni][e] = 0.0f;

  const int ra = tid >> 2;            // 0..31  (A staging row)
  const int ca = (tid & 3) * 8;       // 0,8,16,24 (16B chunks)
  const int rb = tid >> 5;            // 0..3   (B staging row)
  const int cb = (tid & 31) * 4;      // 0..124

  for (int k0 = 0; k0 < K; k0 += BK) {
    // --- stage A tile 128x32: async global->LDS (no VGPR round-trip) ---
#ifdef HAVE_ASYNC_LDS
#pragma unroll
    for (int rr = 0; rr < BM; rr += 32)
      ASYNC_COPY_B128(A + (size_t)(bm0 + ra + rr) * K + k0 + ca,
                      &As[(ra + rr) * LDA + ca]);
#else
    {
      uint4 t[4];
#pragma unroll
      for (int rr = 0; rr < 4; rr++)
        t[rr] = *(const uint4*)(A + (size_t)(bm0 + ra + rr * 32) * K + k0 + ca);
#pragma unroll
      for (int rr = 0; rr < 4; rr++)
        *(uint4*)&As[(ra + rr * 32) * LDA + ca] = t[rr];
    }
#endif
    // --- stage B tile 32x128, TRANSPOSED into Bst[n][k]: load-all, store-all ---
    {
      bf16 tb[8][4];
#pragma unroll
      for (int rr = 0; rr < 8; rr++)
        *(uint2*)tb[rr] = *(const uint2*)(Bw + (size_t)(k0 + rb + rr * 4) * N + bn0 + cb);
#pragma unroll
      for (int rr = 0; rr < 8; rr++)
#pragma unroll
        for (int j = 0; j < 4; j++) Bst[(cb + j) * LDBT + rb + rr * 4] = tb[rr][j];
    }
    // --- prefetch next K-tile while this one is consumed ---
    if (k0 + BK < K) {
      __builtin_prefetch(A + (size_t)(bm0 + ra) * K + k0 + BK + ca, 0, 0);
      __builtin_prefetch(Bw + (size_t)(k0 + BK + rb) * N + bn0 + cb, 0, 0);
    }
#ifdef HAVE_ASYNC_LDS
    ASYNC_WAIT();
#endif
    __syncthreads();

    // --- A fragments: a[i] = A[m][ i%8 + 16*(i/8) + 8*half ] (two b128s) ---
    v16bf af[4];
#pragma unroll
    for (int mi = 0; mi < 4; mi++) {
      const bf16* base = &As[(mo + mi * 16 + lm) * LDA + half * 8];
#pragma unroll
      for (int i = 0; i < 8; i++) af[mi][i] = base[i];
#pragma unroll
      for (int i = 0; i < 8; i++) af[mi][8 + i] = base[16 + i];
    }
    // --- B fragments: b[i] = B[ i%8 + 16*(i/8) + 8*half ][n] (matches A) ---
    v16bf bfv[4];
#pragma unroll
    for (int ni = 0; ni < 4; ni++) {
      const bf16* base = &Bst[(no + ni * 16 + lm) * LDBT + half * 8];
#pragma unroll
      for (int i = 0; i < 16; i++) bfv[ni][i] = base[i + (i >= 8 ? 8 : 0)];
    }
#pragma unroll
    for (int mi = 0; mi < 4; mi++)
#pragma unroll
      for (int ni = 0; ni < 4; ni++)
        acc[mi][ni] = WMMA_BF16(af[mi], bfv[ni], acc[mi][ni]);
    __syncthreads();
  }

  // --- epilogue: C/D layout row = v + 8*half, col = lane%16; branch hoisted ---
  if (Cf) {
#pragma unroll
    for (int mi = 0; mi < 4; mi++) {
#pragma unroll
      for (int ni = 0; ni < 4; ni++) {
        const int gn = bn0 + no + ni * 16 + lm;
        const float bv = bias[gn];
        const int row0 = bm0 + mo + mi * 16 + half * 8;
#pragma unroll
        for (int v = 0; v < 8; v++) Cf[(row0 + v) * N + gn] = acc[mi][ni][v] + bv;
      }
    }
  } else {
#pragma unroll
    for (int mi = 0; mi < 4; mi++) {
#pragma unroll
      for (int ni = 0; ni < 4; ni++) {
        const int gn = bn0 + no + ni * 16 + lm;
        const float bv = bias[gn];
        const int row0 = bm0 + mo + mi * 16 + half * 8;
#pragma unroll
        for (int v = 0; v < 8; v++) Cb[(row0 + v) * N + gn] = (bf16)(acc[mi][ni][v] + bv);
      }
    }
  }
}

// ---------------------------------------------------------------------------
// Flash attention (causal), bf16 WMMA, f32 softmax stats.
// grid = (T/128 q-tiles, B*H). block = 256 (8 waves); wave owns 16 q-rows.
// Faithful to reference: energy scaled by 1/sqrt(E) AFTER adding mask*(-1e9).
// ---------------------------------------------------------------------------
__global__ __launch_bounds__(256) void flash_attn_bf16(const bf16* __restrict__ Q,
                                                       const bf16* __restrict__ Kk,
                                                       const bf16* __restrict__ V,
                                                       bf16* __restrict__ O,
                                                       int T, float scale, float negScaled) {
  constexpr int D = 64, BQ = 128, BKV = 128, E = 1024, H = 16;
  constexpr int LDK = 72;   // Ks [128][72] (144 B rows, 16B aligned)
  constexpr int LDV = 136;  // Vt [64][136]  (shares buffer with Ks, phased)
  constexpr int LDP = 136;  // Ps per-wave [16][136]
  __shared__ __align__(16) bf16 kv[BKV * LDK];       // 18432 B
  __shared__ __align__(16) bf16 Ps[8 * 16 * LDP];    // 34816 B

  const int tid  = threadIdx.x;
  const int lane = tid & 31, wave = tid >> 5;
  const int half = lane >> 4, lm = lane & 15;
  const int qt = blockIdx.x;
  const int bh = blockIdx.y;
  const int b = bh >> 4, h = bh & (H - 1);
  const size_t base = (size_t)b * T * E + (size_t)h * D;  // token stride = E

  // Q fragments for this wave's 16 rows (kept in registers for the k-loop)
  const int qrow = qt * BQ + wave * 16 + lm;
  v16bf aq[2];
#pragma unroll
  for (int ks = 0; ks < 2; ks++) {
    const bf16* qp = Q + base + (size_t)qrow * E + ks * 32 + half * 8;
#pragma unroll
    for (int i = 0; i < 8; i++) aq[ks][i] = qp[i];
#pragma unroll
    for (int i = 0; i < 8; i++) aq[ks][8 + i] = qp[16 + i];
  }

  float mrow[8], lrow[8];
  v8f o[4];
#pragma unroll
  for (int v = 0; v < 8; v++) { mrow[v] = -3.0e38f; lrow[v] = 0.0f; }
#pragma unroll
  for (int dn = 0; dn < 4; dn++)
#pragma unroll
    for (int v = 0; v < 8; v++) o[dn][v] = 0.0f;

  bf16* Ks = kv;                 // phase A: K tile [128][LDK]
  bf16* Vt = kv;                 // phase B: V^T tile [64][LDV]
  bf16* Pw = Ps + wave * 16 * LDP;

  const int rkv = tid >> 2;            // 0..63  (K/V staging row)
  const int seg = (tid & 3) * 16;      // 0,16,32,48

  for (int kt = 0; kt <= qt; kt++) {   // causal: key tiles <= query tile
    __syncthreads();
    // ---- stage K tile [128 keys][64 d]: async global->LDS ----
#ifdef HAVE_ASYNC_LDS
#pragma unroll
    for (int rr = 0; rr < BKV; rr += 64) {
      const bf16* s = Kk + base + (size_t)(kt * BKV + rkv + rr) * E + seg;
      bf16* l = &Ks[(rkv + rr) * LDK + seg];
      ASYNC_COPY_B128(s, l);
      ASYNC_COPY_B128(s + 8, l + 8);
    }
#else
    {
      uint4 t[4];
#pragma unroll
      for (int rr = 0; rr < 2; rr++) {
        const bf16* s = Kk + base + (size_t)(kt * BKV + rkv + rr * 64) * E + seg;
        t[rr * 2 + 0] = *(const uint4*)s;
        t[rr * 2 + 1] = *(const uint4*)(s + 8);
      }
#pragma unroll
      for (int rr = 0; rr < 2; rr++) {
        bf16* l = &Ks[(rkv + rr * 64) * LDK + seg];
        *(uint4*)l = t[rr * 2 + 0];
        *(uint4*)(l + 8) = t[rr * 2 + 1];
      }
    }
#endif
    // prefetch the matching V tile lines while S is computed
    __builtin_prefetch(V + base + (size_t)(kt * BKV + rkv) * E + seg, 0, 0);
#ifdef HAVE_ASYNC_LDS
    ASYNC_WAIT();
#endif
    __syncthreads();

    // ---- S = Q * K^T : 8 key subtiles, K-dim = D = 64 -> 2 wmma steps ----
    v8f s8[8];
#pragma unroll
    for (int n = 0; n < 8; n++)
#pragma unroll
      for (int e = 0; e < 8; e++) s8[n][e] = 0.0f;
#pragma unroll
    for (int ks = 0; ks < 2; ks++) {
#pragma unroll
      for (int n = 0; n < 8; n++) {
        v16bf bk;
        const bf16* kp = &Ks[(n * 16 + lm) * LDK + ks * 32 + half * 8];
#pragma unroll
        for (int i = 0; i < 16; i++) bk[i] = kp[i + (i >= 8 ? 8 : 0)];
        s8[n] = WMMA_BF16(aq[ks], bk, s8[n]);
      }
    }

    // ---- scale + causal mask (matches reference: S/sqrt(E) + mask*NEG/sqrt(E)) ----
    const int qg0 = qt * BQ + wave * 16 + half * 8;  // + v = this lane's row ids
#pragma unroll
    for (int n = 0; n < 8; n++) {
      int kg = kt * BKV + n * 16 + lm;
#pragma unroll
      for (int v = 0; v < 8; v++) {
        float e = s8[n][v] * scale;
        if (kg > qg0 + v) e += negScaled;
        s8[n][v] = e;
      }
    }

    // ---- online softmax: row reductions across the 16-lane half-group ----
    float alpha[8], rmax[8], rsum[8];
#pragma unroll
    for (int v = 0; v < 8; v++) {
      float m = -3.0e38f;
#pragma unroll
      for (int n = 0; n < 8; n++) m = fmaxf(m, s8[n][v]);
      m = fmaxf(m, __shfl_xor(m, 1, 32));
      m = fmaxf(m, __shfl_xor(m, 2, 32));
      m = fmaxf(m, __shfl_xor(m, 4, 32));
      m = fmaxf(m, __shfl_xor(m, 8, 32));
      float mnew = fmaxf(mrow[v], m);
      alpha[v] = __expf(mrow[v] - mnew);
      mrow[v] = mnew;
      rmax[v] = mnew;
      rsum[v] = 0.0f;
    }
    // exp, accumulate row sums, stage P (bf16) for the P*V wmma
#pragma unroll
    for (int n = 0; n < 8; n++) {
#pragma unroll
      for (int v = 0; v < 8; v++) {
        float pe = __expf(s8[n][v] - rmax[v]);
        rsum[v] += pe;
        Pw[(half * 8 + v) * LDP + n * 16 + lm] = (bf16)pe;
      }
    }
#pragma unroll
    for (int v = 0; v < 8; v++) {
      float rs = rsum[v];
      rs += __shfl_xor(rs, 1, 32);
      rs += __shfl_xor(rs, 2, 32);
      rs += __shfl_xor(rs, 4, 32);
      rs += __shfl_xor(rs, 8, 32);
      lrow[v] = lrow[v] * alpha[v] + rs;
    }
    // rescale O accumulators by alpha
#pragma unroll
    for (int dn = 0; dn < 4; dn++)
#pragma unroll
      for (int v = 0; v < 8; v++) o[dn][v] *= alpha[v];

    __syncthreads();
    // ---- stage V tile transposed: Vt[d][kv]; load-all then store-all ----
    {
      bf16 tv[2][16];
#pragma unroll
      for (int rr = 0; rr < 2; rr++) {
        const bf16* s = V + base + (size_t)(kt * BKV + rkv + rr * 64) * E + seg;
        *(uint4*)&tv[rr][0] = *(const uint4*)s;
        *(uint4*)&tv[rr][8] = *(const uint4*)(s + 8);
      }
#pragma unroll
      for (int rr = 0; rr < 2; rr++)
#pragma unroll
        for (int i = 0; i < 16; i++) Vt[(seg + i) * LDV + rkv + rr * 64] = tv[rr][i];
    }
    __syncthreads();

    // ---- O += P * V : K-dim = 128 keys -> 4 wmma steps, 4 d subtiles ----
#pragma unroll
    for (int ks = 0; ks < 4; ks++) {
      v16bf ap;
      const bf16* pp = &Pw[lm * LDP + ks * 32 + half * 8];
#pragma unroll
      for (int i = 0; i < 8; i++) ap[i] = pp[i];
#pragma unroll
      for (int i = 0; i < 8; i++) ap[8 + i] = pp[16 + i];
#pragma unroll
      for (int dn = 0; dn < 4; dn++) {
        v16bf bv;
        const bf16* vp = &Vt[(dn * 16 + lm) * LDV + ks * 32 + half * 8];
#pragma unroll
        for (int i = 0; i < 16; i++) bv[i] = vp[i + (i >= 8 ? 8 : 0)];
        o[dn] = WMMA_BF16(ap, bv, o[dn]);
      }
    }
  }

  // ---- normalize and write attention output (bf16, [B,T,H,D] layout) ----
#pragma unroll
  for (int dn = 0; dn < 4; dn++) {
#pragma unroll
    for (int v = 0; v < 8; v++) {
      int tok = qt * BQ + wave * 16 + half * 8 + v;
      float val = o[dn][v] / lrow[v];
      O[base + (size_t)tok * E + dn * 16 + lm] = (bf16)val;
    }
  }
}

// ---------------------------------------------------------------------------
// Host-side orchestration
// ---------------------------------------------------------------------------
extern "C" void kernel_launch(void* const* d_in, const int* in_sizes, int n_in,
                              void* d_out, int out_size, void* d_ws, size_t ws_size,
                              hipStream_t stream) {
  const float* x  = (const float*)d_in[0];
  const float* Wq = (const float*)d_in[1];
  const float* bq = (const float*)d_in[2];
  const float* Wk = (const float*)d_in[3];
  const float* bk = (const float*)d_in[4];
  const float* Wv = (const float*)d_in[5];
  const float* bv = (const float*)d_in[6];
  const float* Wo = (const float*)d_in[7];
  const float* bo = (const float*)d_in[8];
  float* out = (float*)d_out;

  const int Bb = 2, T = 2048, E = 1024, H = 16;
  const int M = Bb * T;  // 4096 tokens

  const size_t MiB = 1024 * 1024;
  char* ws = (char*)d_ws;
  bf16* xb  = (bf16*)(ws + 0);          // 8 MiB; reused as attention output later
  bf16* wqb = (bf16*)(ws + 8  * MiB);   // 2 MiB each
  bf16* wkb = (bf16*)(ws + 10 * MiB);
  bf16* wvb = (bf16*)(ws + 12 * MiB);
  bf16* wob = (bf16*)(ws + 14 * MiB);
  bf16* Qb  = (bf16*)(ws + 16 * MiB);   // 8 MiB each
  bf16* Kb  = (bf16*)(ws + 24 * MiB);
  bf16* Vb  = (bf16*)(ws + 32 * MiB);   // total 40 MiB
  bf16* Ab  = xb;                       // x no longer needed after projections

  const int nX4 = (M * E) / 4;  // 1,048,576 quads
  const int nW4 = (E * E) / 4;  //   262,144 quads
  cvt_f32_to_bf16_x4<<<(nX4 + 255) / 256, 256, 0, stream>>>(x,  xb,  nX4);
  cvt_f32_to_bf16_x4<<<(nW4 + 255) / 256, 256, 0, stream>>>(Wq, wqb, nW4);
  cvt_f32_to_bf16_x4<<<(nW4 + 255) / 256, 256, 0, stream>>>(Wk, wkb, nW4);
  cvt_f32_to_bf16_x4<<<(nW4 + 255) / 256, 256, 0, stream>>>(Wv, wvb, nW4);
  cvt_f32_to_bf16_x4<<<(nW4 + 255) / 256, 256, 0, stream>>>(Wo, wob, nW4);

  dim3 ggrid(E / 128, M / 128);  // (8, 32)
  dim3 gblk(128);                // 4 waves, 2x2 wave grid
  gemm_wmma_bf16<<<ggrid, gblk, 0, stream>>>(xb, wqb, bq, nullptr, Qb, M, E, E);
  gemm_wmma_bf16<<<ggrid, gblk, 0, stream>>>(xb, wkb, bk, nullptr, Kb, M, E, E);
  gemm_wmma_bf16<<<ggrid, gblk, 0, stream>>>(xb, wvb, bv, nullptr, Vb, M, E, E);

  const float scale = 0.03125f;              // 1/sqrt(1024)
  const float negScaled = -1.0e9f * scale;   // mask added before the sqrt(E) scaling
  dim3 fgrid(T / 128, Bb * H);  // (16, 32)
  flash_attn_bf16<<<fgrid, dim3(256), 0, stream>>>(Qb, Kb, Vb, Ab, T, scale, negScaled);

  gemm_wmma_bf16<<<ggrid, gblk, 0, stream>>>(Ab, wob, bo, out, nullptr, M, E, E);
}
